// HierarchicalSparseFieldClassifier_69982197121715
// MI455X (gfx1250) — compile-verified, matrix operand order checked by reading, not run on hardware
//
#include <hip/hip_runtime.h>
#include <hip/hip_bf16.h>
#include <math.h>

typedef __attribute__((ext_vector_type(16))) _Float16 v16h;
typedef __attribute__((ext_vector_type(8)))  float    v8f;

#define NB    64
#define TLEN  2048
#define DMD   128
#define CDD   32
#define NCH   512     /* T / CH */
#define NCLS  8
#define FEATN 418
#define LAMBDA 0.05f
#define RHOF   0.9f
#define LRF    0.03f
#define FLT_BIG 3.402823466e38f

// ---------- wave32 helpers ----------
__device__ __forceinline__ float wred32(float v) {
#pragma unroll
  for (int o = 16; o; o >>= 1) v += __shfl_xor(v, o, 32);
  return v;
}

// CDNA5 16-bit A/B fragment K index for half h (0..15) of a lane
__device__ __forceinline__ int frag_koff(int h, int lane) {
  return ((h < 8) ? 0 : 16) + ((lane < 16) ? 0 : 8) + (h & 7);
}

__device__ __forceinline__ v8f wmma_f16(v16h a, v16h b, v8f c) {
  return __builtin_amdgcn_wmma_f32_16x16x32_f16(false, a, false, b, (short)0, c, false, false);
}

// ---------- one-time weight swizzle: W (Ktot x N, f32) -> f16 WMMA B-fragments ----------
// layout: [(ntile*nkc + kc)*32 + lane][16 halves]; one v16h per lane per (tile,kstep)
__global__ void k_prep(const float* __restrict__ W, _Float16* __restrict__ out, int Ktot, int N) {
  const int idx = blockIdx.x * blockDim.x + threadIdx.x;
  const int nkc = Ktot >> 5, ntiles = N >> 4;
  if (idx >= ntiles * nkc * 32) return;
  const int lane = idx & 31;
  const int kc   = (idx >> 5) % nkc;
  const int nt   = (idx >> 5) / nkc;
  const int n    = nt * 16 + (lane & 15);
  _Float16* o = out + (size_t)idx * 16;
#pragma unroll
  for (int h = 0; h < 16; ++h) {
    const int k = kc * 32 + frag_koff(h, lane);
    o[h] = (_Float16)W[(size_t)k * N + n];
  }
}

// ---------- init: column-normalized D0 per batch, c0 = 0 ----------
__global__ __launch_bounds__(256) void k_init(const float* __restrict__ base_D,
                                              float* __restrict__ Dg, float* __restrict__ cg) {
  const int b = blockIdx.x, tid = threadIdx.x;
  __shared__ float nrm[CDD];
  if (tid < CDD) {
    float s = 0.f;
    for (int dm = 0; dm < DMD; ++dm) { float v = base_D[dm * CDD + tid]; s += v * v; }
    nrm[tid] = fmaxf(sqrtf(s), 1e-12f);
  }
  __syncthreads();
  for (int q = tid; q < DMD * CDD; q += 256)
    Dg[(size_t)b * DMD * CDD + q] = base_D[q] / nrm[q & 31];
  if (tid < CDD) cg[b * CDD + tid] = 0.f;
}

// ---------- fused embed + LN + convs(GLU) + proj + LN + chunk stats ----------
__global__ __launch_bounds__(256) void k_encode(
    const int* __restrict__ tokens, const float* __restrict__ emb,
    const float* __restrict__ in_w, const float* __restrict__ in_b,
    const _Float16* __restrict__ c1F, const float* __restrict__ c1b,
    const _Float16* __restrict__ c2F, const float* __restrict__ c2b,
    const _Float16* __restrict__ opF, const float* __restrict__ op_b,
    const float* __restrict__ on_w, const float* __restrict__ on_b,
    float* __restrict__ chunk_h, float* __restrict__ chunk_cnt,
    float* __restrict__ chunk_mask, float* __restrict__ chunk_max) {
  __shared__ float xf32[20][DMD];
  __shared__ v16h  fragv[640];           // conv A-frags (5 shifts x 4 kc x 32 lanes); reused for proj A-frags
  __shared__ float pre[16][512];         // conv pre-GLU; later reused as h buffer
  __shared__ int   tokm[20];
  _Float16* fragh = (_Float16*)fragv;
  float (*hbuf)[DMD] = reinterpret_cast<float(*)[DMD]>(&pre[0][0]);

  const int b    = blockIdx.x >> 7;      // 128 tiles per batch
  const int tile = blockIdx.x & 127;
  const int t0   = tile << 4;
  const int tid  = threadIdx.x;
  const int wave = tid >> 5;
  const int lane = tid & 31;

  // ---- embedding + input LayerNorm, rows t0-2 .. t0+17 (conv halo) ----
  for (int p = wave; p < 20; p += 8) {
    const int pg  = t0 - 2 + p;
    const int tok = (pg >= 0 && pg < TLEN) ? tokens[b * TLEN + pg] : -1;
    if (lane == 0) tokm[p] = tok;
    if (tok >= 0) {
      float e[4], s = 0.f;
#pragma unroll
      for (int i = 0; i < 4; ++i) { e[i] = emb[(size_t)tok * DMD + lane + 32 * i]; s += e[i]; }
      s = wred32(s);
      const float mean = s * (1.f / DMD);
      float vs = 0.f;
#pragma unroll
      for (int i = 0; i < 4; ++i) { float d = e[i] - mean; vs += d * d; }
      vs = wred32(vs);
      const float rs = rsqrtf(vs * (1.f / DMD) + 1e-5f);
#pragma unroll
      for (int i = 0; i < 4; ++i) {
        const int dm = lane + 32 * i;
        xf32[p][dm] = (e[i] - mean) * rs * in_w[dm] + in_b[dm];
      }
    } else {
#pragma unroll
      for (int i = 0; i < 4; ++i) xf32[p][lane + 32 * i] = 0.f;
    }
  }
  __syncthreads();

  // ---- build conv A-fragments: 5 halo shifts x 4 K-chunks, f16 fragment order ----
  for (int q = tid; q < 5 * 4 * 32 * 16; q += 256) {
    const int h = q & 15, ln = (q >> 4) & 31, kc = (q >> 9) & 3, r0 = q >> 11;
    fragh[q] = (_Float16)xf32[(ln & 15) + r0][kc * 32 + frag_koff(h, ln)];
  }
  __syncthreads();

  // ---- conv1 (3 taps) + conv2 (5 taps), pre-GLU, via WMMA ----
  for (int j = wave * 4; j < wave * 4 + 4; ++j) {   // 32 N-tiles, 4 per wave
    const bool is1 = j < 16;
    const int nt   = is1 ? j : j - 16;
    const v16h* Wv = (const v16h*)(is1 ? c1F : c2F);
    const float* bias = is1 ? c1b : c2b;
    const int taps  = is1 ? 3 : 5;
    const int shift = is1 ? 1 : 0;
    const int nkc   = taps * 4;
    v8f acc = {};
    for (int tap = 0; tap < taps; ++tap) {
      for (int kc = 0; kc < 4; ++kc) {
        v16h af = fragv[(((tap + shift) * 4 + kc) * 32) + lane];
        v16h bf = Wv[((size_t)nt * nkc + tap * 4 + kc) * 32 + lane];
        acc = wmma_f16(af, bf, acc);
      }
    }
    const int n = lane & 15, mo = (lane < 16) ? 0 : 8, col = nt * 16 + n;
#pragma unroll
    for (int v = 0; v < 8; ++v) pre[v + mo][(is1 ? 0 : 256) + col] = acc[v] + bias[col];
  }
  __syncthreads();

  // ---- GLU fused into projection A-fragment construction (reuses fragv) ----
  for (int q = tid; q < 8 * 32 * 16; q += 256) {
    const int h = q & 15, ln = (q >> 4) & 31, kc = q >> 9;
    const int t = ln & 15, jj = kc * 32 + frag_koff(h, ln);
    float av, gv;
    if (jj < 128) { av = pre[t][jj];       gv = pre[t][jj + 128]; }
    else          { av = pre[t][jj + 128]; gv = pre[t][jj + 256]; }
    fragh[q] = (_Float16)(av * (1.f / (1.f + expf(-gv))));
  }
  __syncthreads();

  // ---- output projection (16x256 @ 256x128) + bias + residual (aliases pre) ----
  {
    const int nt = wave;                 // 8 N-tiles, one per wave
    const v16h* Wv = (const v16h*)opF;
    v8f acc = {};
    for (int kc = 0; kc < 8; ++kc)
      acc = wmma_f16(fragv[kc * 32 + lane], Wv[((size_t)nt * 8 + kc) * 32 + lane], acc);
    const int n = lane & 15, mo = (lane < 16) ? 0 : 8, col = nt * 16 + n;
#pragma unroll
    for (int v = 0; v < 8; ++v) {
      const int t = v + mo;
      hbuf[t][col] = acc[v] + op_b[col] + xf32[t + 2][col];
    }
  }
  __syncthreads();

  // ---- LayerNorm h (in place; wave handles rows wave, wave+8) ----
  for (int rr = 0; rr < 2; ++rr) {
    const int t = wave + rr * 8;
    float e[4], s = 0.f;
#pragma unroll
    for (int i = 0; i < 4; ++i) { e[i] = hbuf[t][lane + 32 * i]; s += e[i]; }
    s = wred32(s);
    const float mean = s * (1.f / DMD);
    float vs = 0.f;
#pragma unroll
    for (int i = 0; i < 4; ++i) { float d = e[i] - mean; vs += d * d; }
    vs = wred32(vs);
    const float rs = rsqrtf(vs * (1.f / DMD) + 1e-5f);
#pragma unroll
    for (int i = 0; i < 4; ++i) {
      const int dm = lane + 32 * i;
      hbuf[t][dm] = (e[i] - mean) * rs * on_w[dm] + on_b[dm];
    }
  }
  __syncthreads();

  // ---- chunk stats (4 chunks of 4 timesteps per tile) ----
  for (int q = tid; q < 4 * DMD; q += 256) {
    const int c = q >> 7, dm = q & 127;
    float s = 0.f, mx = -FLT_BIG; int cnt = 0;
#pragma unroll
    for (int i = 0; i < 4; ++i) {
      const int t = c * 4 + i;
      if (tokm[t + 2] != 0) { const float hv = hbuf[t][dm]; s += hv; mx = fmaxf(mx, hv); ++cnt; }
    }
    const size_t ci = (size_t)b * NCH + (tile * 4 + c);
    chunk_h[ci * DMD + dm]   = s / fmaxf((float)cnt, 1.f);
    chunk_max[ci * DMD + dm] = mx;
    if (dm == 0) { chunk_cnt[ci] = (float)cnt; chunk_mask[ci] = (cnt > 0) ? 1.f : 0.f; }
  }
}

// ---------- fold chunks -> surface mean / max (deterministic) ----------
__global__ void k_surface(const float* __restrict__ chunk_h, const float* __restrict__ chunk_cnt,
                          const float* __restrict__ chunk_max,
                          float* __restrict__ smean, float* __restrict__ smax) {
  const int i = blockIdx.x * blockDim.x + threadIdx.x;
  if (i >= NB * DMD) return;
  const int b = i >> 7, dm = i & 127;
  float tot = 0.f, totc = 0.f, mx = -FLT_BIG;
  for (int c = 0; c < NCH; ++c) {
    const size_t ci = (size_t)b * NCH + c;
    const float cnt = chunk_cnt[ci];
    tot  += chunk_h[ci * DMD + dm] * cnt;
    mx    = fmaxf(mx, chunk_max[ci * DMD + dm]);
    totc += cnt;
  }
  smean[i] = tot / fmaxf(totc, 1.f);
  smax[i]  = (totc > 0.f) ? mx : 0.f;
}

// ---------- 512-step sequential dictionary scan; 4 WGs x 16 batches ----------
__global__ __launch_bounds__(256) void k_scan(
    const float* __restrict__ chunk_h, const float* __restrict__ chunk_mask,
    const float* __restrict__ wcu, const float* __restrict__ wcc,
    const _Float16* __restrict__ wcduF, const float* __restrict__ bcdu,
    const _Float16* __restrict__ wcdcF, const float* __restrict__ bcdc,
    const float* __restrict__ wg, const float* __restrict__ bg,
    const float* __restrict__ un_w, const float* __restrict__ un_b,
    const float* __restrict__ cn_w, const float* __restrict__ cn_b,
    float* __restrict__ Dg, float* __restrict__ cg, float* __restrict__ cand_ws,
    float* __restrict__ z_out, float* __restrict__ err_out, float* __restrict__ gate_out) {
  const int b0 = blockIdx.x * 16;
  float* cand = cand_ws + (size_t)blockIdx.x * 16 * 4096;
  const int tid = threadIdx.x, wave = tid >> 5, lane = tid & 31;

  __shared__ float uS[16][DMD];
  __shared__ v16h  uA[4 * 32];           // u A-fragments  (4 kc x 32 lanes)
  __shared__ v16h  ctA[32];              // c_t A-fragments (1 kc x 32 lanes)
  __shared__ float unS[16][DMD];
  __shared__ float rS[16][DMD];
  __shared__ float cnS[16][CDD];
  __shared__ float lgS[16][CDD];
  __shared__ float aS[16][CDD];
  __shared__ float ctS[16][CDD];
  __shared__ float errP[16][16];
  __shared__ float errS[16], gateS[16], validS[16];
  _Float16* uAh  = (_Float16*)uA;
  _Float16* ctAh = (_Float16*)ctA;

  for (int s = 0; s < NCH; ++s) {
    for (int q = tid; q < 16 * DMD; q += 256) {
      const int rb = q >> 7, dm = q & 127;
      uS[rb][dm] = chunk_h[(((size_t)(b0 + rb)) * NCH + s) * DMD + dm];
    }
    if (tid < 16) validS[tid] = chunk_mask[(size_t)(b0 + tid) * NCH + s];
    if (s + 1 < NCH)   // hide next step's u fetch behind this step's math
      __builtin_prefetch(&chunk_h[(((size_t)(b0 + (tid >> 4))) * NCH + s + 1) * DMD + (tid & 15) * 8], 0, 1);
    __syncthreads();

    // u A-fragments
    for (int q = tid; q < 4 * 32 * 16; q += 256) {
      const int h = q & 15, ln = (q >> 4) & 31, kc = q >> 9;
      uAh[q] = (_Float16)uS[ln & 15][kc * 32 + frag_koff(h, ln)];
    }
    // LN(u) and LN(c_prev); wave handles rows wave, wave+8
    for (int rr = 0; rr < 2; ++rr) {
      const int rb = wave + rr * 8;
      float e[4], sm = 0.f;
#pragma unroll
      for (int i = 0; i < 4; ++i) { e[i] = uS[rb][lane + 32 * i]; sm += e[i]; }
      sm = wred32(sm);
      const float mean = sm * (1.f / DMD);
      float vs = 0.f;
#pragma unroll
      for (int i = 0; i < 4; ++i) { float d = e[i] - mean; vs += d * d; }
      vs = wred32(vs);
      const float rs = rsqrtf(vs * (1.f / DMD) + 1e-5f);
#pragma unroll
      for (int i = 0; i < 4; ++i) {
        const int dm = lane + 32 * i;
        unS[rb][dm] = (e[i] - mean) * rs * un_w[dm] + un_b[dm];
      }
      const float cv = cg[(b0 + rb) * CDD + lane];
      const float cm = wred32(cv) * (1.f / CDD);
      const float cd = cv - cm;
      const float cvr = wred32(cd * cd) * (1.f / CDD);
      cnS[rb][lane] = cd * rsqrtf(cvr + 1e-5f) * cn_w[lane] + cn_b[lane];
    }
    __syncthreads();

    // logits = ln(u)@wcu + ln(c)@wcc
    for (int q = tid; q < 16 * CDD; q += 256) {
      const int rb = q >> 5, cd = q & 31;
      float acc = 0.f;
      for (int dm = 0; dm < DMD; ++dm) acc += unS[rb][dm] * wcu[dm * CDD + cd];
      for (int j = 0; j < CDD; ++j)    acc += cnS[rb][j] * wcc[j * CDD + cd];
      lgS[rb][cd] = acc;
    }
    __syncthreads();

    // soft-threshold + wave32 rank-based top-8; c update
    for (int rr = 0; rr < 2; ++rr) {
      const int rb = wave + rr * 8;
      const float x  = lgS[rb][lane];
      const float ax = fabsf(x);
      const float sv = (ax > LAMBDA) ? copysignf(ax - LAMBDA, x) : 0.f;
      const float asv = fabsf(sv);
      int rank = 0;
      for (int j = 0; j < 32; ++j) {
        const float o = __shfl(asv, j, 32);
        rank += (o > asv) || (o == asv && j < lane);
      }
      const float a = (rank < 8) ? sv : 0.f;
      aS[rb][lane] = a;
      const float cp   = cg[(b0 + rb) * CDD + lane];
      const float cnew = RHOF * cp + (1.f - RHOF) * a;
      const float v    = validS[rb];
      ctS[rb][lane] = v * cnew + (1.f - v) * cp;
    }
    __syncthreads();

    // c_t A-fragments + u_hat / residual / err partials
    for (int q = tid; q < 32 * 16; q += 256) {
      const int h = q & 15, ln = q >> 4;
      ctAh[q] = (_Float16)ctS[ln & 15][frag_koff(h, ln)];
    }
    {
      const int rb = tid >> 4, g8 = (tid & 15) * 8;
      float ep = 0.f;
      const float* Drow0 = Dg + ((size_t)(b0 + rb) * DMD) * CDD;
      for (int i = 0; i < 8; ++i) {
        const int dm = g8 + i;
        float acc = 0.f;
        const float* Dr = Drow0 + (size_t)dm * CDD;
        for (int cd = 0; cd < CDD; ++cd) acc += Dr[cd] * aS[rb][cd];
        const float rv = uS[rb][dm] - acc;
        rS[rb][dm] = rv; ep += rv * rv;
      }
      errP[rb][tid & 15] = ep;
    }
    __syncthreads();
    if (tid < 16) {
      float e = 0.f;
      for (int i = 0; i < 16; ++i) e += errP[tid][i];
      errS[tid] = sqrtf(e);
    }
    __syncthreads();
    if (tid < 16) {
      float acc = bg[0];
      for (int dm = 0; dm < DMD; ++dm) acc += uS[tid][dm] * wg[dm];
      for (int j = 0; j < CDD; ++j)    acc += ctS[tid][j] * wg[DMD + j];
      acc += errS[tid] * wg[DMD + CDD];
      gateS[tid] = 1.f / (1.f + expf(-acc));
    }

    // cand = u@wcdu + c_t@wcdc via WMMA (M=16, N=4096, K=128+32); biases folded later
    {
      const v16h* WduV = (const v16h*)wcduF;   // [nt*4+kc][32 lanes]
      const v16h* WdcV = (const v16h*)wcdcF;   // [nt][32 lanes]
      for (int i = 0; i < 32; ++i) {
        const int nt = wave + 8 * i;
        v8f acc = {};
#pragma unroll
        for (int kc = 0; kc < 4; ++kc)
          acc = wmma_f16(uA[kc * 32 + lane], WduV[((size_t)nt * 4 + kc) * 32 + lane], acc);
        acc = wmma_f16(ctA[lane], WdcV[(size_t)nt * 32 + lane], acc);
        const int n = lane & 15, mo = (lane < 16) ? 0 : 8;
#pragma unroll
        for (int v = 0; v < 8; ++v) cand[(size_t)(v + mo) * 4096 + nt * 16 + n] = acc[v];
      }
    }
    __threadfence_block();
    __syncthreads();

    // dictionary update: each thread owns columns (batch-row, cd)
    for (int q = tid; q < 16 * CDD; q += 256) {
      const int rb = q >> 5, cd = q & 31;
      const float g = gateS[rb], av = aS[rb][cd], v = validS[rb];
      float* Dcol = Dg + ((size_t)(b0 + rb) * DMD) * CDD + cd;
      const float* crow = cand + (size_t)rb * 4096;
      float n1 = 0.f, n2 = 0.f;
      for (int dm = 0; dm < DMD; ++dm) {
        const float t1 = Dcol[dm * CDD] + LRF * rS[rb][dm] * av;
        n1 += t1 * t1;
        const int ci = dm * CDD + cd;
        const float cvv = crow[ci] + bcdu[ci] + bcdc[ci];
        n2 += cvv * cvv;
      }
      n1 = fmaxf(sqrtf(n1), 1e-12f);
      n2 = fmaxf(sqrtf(n2), 1e-12f);
      float n3 = 0.f;
      for (int dm = 0; dm < DMD; ++dm) {
        const float t1 = Dcol[dm * CDD] + LRF * rS[rb][dm] * av;
        const int ci = dm * CDD + cd;
        const float cvv = crow[ci] + bcdu[ci] + bcdc[ci];
        const float dn = (1.f - g) * (t1 / n1) + g * (cvv / n2);
        n3 += dn * dn;
      }
      n3 = fmaxf(sqrtf(n3), 1e-12f);
      if (v != 0.f) {
        for (int dm = 0; dm < DMD; ++dm) {
          const float t1 = Dcol[dm * CDD] + LRF * rS[rb][dm] * av;
          const int ci = dm * CDD + cd;
          const float cvv = crow[ci] + bcdu[ci] + bcdc[ci];
          const float dn = (1.f - g) * (t1 / n1) + g * (cvv / n2);
          Dcol[dm * CDD] = dn / n3;
        }
      }
    }
    __threadfence_block();
    __syncthreads();

    // carry c; final-step outputs (only zs/errs/gates[-1] are needed)
    for (int q = tid; q < 16 * CDD; q += 256) {
      const int rb = q >> 5, cd = q & 31;
      cg[(b0 + rb) * CDD + cd] = ctS[rb][cd];
    }
    if (s == NCH - 1) {
      for (int q = tid; q < 16 * DMD; q += 256) {
        const int rb = q >> 7, dm = q & 127;
        float acc = 0.f;
        const float* Dr = Dg + ((size_t)(b0 + rb) * DMD + dm) * CDD;
        for (int cd = 0; cd < CDD; ++cd) acc += Dr[cd] * ctS[rb][cd];
        z_out[(size_t)(b0 + rb) * DMD + dm] = acc;
      }
      if (tid < 16) { err_out[b0 + tid] = errS[tid]; gate_out[b0 + tid] = gateS[tid]; }
    }
    __threadfence_block();
    __syncthreads();
  }
}

// ---------- classifier head ----------
__global__ __launch_bounds__(256) void k_head(
    const float* __restrict__ z, const float* __restrict__ cg,
    const float* __restrict__ smean, const float* __restrict__ smax,
    const float* __restrict__ err, const float* __restrict__ gate,
    const float* __restrict__ cl_w, const float* __restrict__ cl_b,
    const float* __restrict__ f1w, const float* __restrict__ f1b,
    const float* __restrict__ f2w, const float* __restrict__ f2b,
    float* __restrict__ out) {
  const int b = blockIdx.x, tid = threadIdx.x;
  __shared__ float feat[FEATN];
  __shared__ float featn[FEATN];
  __shared__ float hc[256];
  __shared__ float red[256];
  for (int q = tid; q < FEATN; q += 256) {
    float v;
    if (q < 128)      v = z[b * 128 + q];
    else if (q < 160) v = cg[b * 32 + (q - 128)];
    else if (q < 288) v = smean[b * 128 + (q - 160)];
    else if (q < 416) v = smax[b * 128 + (q - 288)];
    else if (q == 416) v = err[b];
    else               v = gate[b];
    feat[q] = v;
  }
  __syncthreads();
  float p = 0.f;
  for (int q = tid; q < FEATN; q += 256) p += feat[q];
  red[tid] = p; __syncthreads();
  for (int st = 128; st; st >>= 1) { if (tid < st) red[tid] += red[tid + st]; __syncthreads(); }
  const float mean = red[0] / (float)FEATN;
  __syncthreads();
  p = 0.f;
  for (int q = tid; q < FEATN; q += 256) { const float d = feat[q] - mean; p += d * d; }
  red[tid] = p; __syncthreads();
  for (int st = 128; st; st >>= 1) { if (tid < st) red[tid] += red[tid + st]; __syncthreads(); }
  const float rs = rsqrtf(red[0] / (float)FEATN + 1e-5f);
  __syncthreads();
  for (int q = tid; q < FEATN; q += 256) featn[q] = (feat[q] - mean) * rs * cl_w[q] + cl_b[q];
  __syncthreads();
  {
    float x = f1b[tid];
    for (int q = 0; q < FEATN; ++q) x += featn[q] * f1w[(size_t)q * 256 + tid];
    hc[tid] = 0.5f * x * (1.f + erff(x * 0.70710678118654752f));  // exact GELU
  }
  __syncthreads();
  if (tid < NCLS) {
    float acc = f2b[tid];
    for (int j = 0; j < 256; ++j) acc += hc[j] * f2w[j * NCLS + tid];
    out[b * NCLS + tid] = acc;
  }
}

extern "C" void kernel_launch(void* const* d_in, const int* in_sizes, int n_in,
                              void* d_out, int out_size, void* d_ws, size_t ws_size,
                              hipStream_t stream) {
  (void)in_sizes; (void)n_in; (void)out_size; (void)ws_size;
  const int*   tokens = (const int*)  d_in[0];
  const float* emb    = (const float*)d_in[1];
  const float* in_w   = (const float*)d_in[2];
  const float* in_b   = (const float*)d_in[3];
  const float* c1w    = (const float*)d_in[4];
  const float* c1b    = (const float*)d_in[5];
  const float* c2w    = (const float*)d_in[6];
  const float* c2b    = (const float*)d_in[7];
  const float* op_w   = (const float*)d_in[8];
  const float* op_b   = (const float*)d_in[9];
  const float* on_w   = (const float*)d_in[10];
  const float* on_b   = (const float*)d_in[11];
  const float* base_D = (const float*)d_in[12];
  const float* wcu    = (const float*)d_in[13];
  const float* wcc    = (const float*)d_in[14];
  const float* wcdu   = (const float*)d_in[15];
  const float* bcdu   = (const float*)d_in[16];
  const float* wcdc   = (const float*)d_in[17];
  const float* bcdc   = (const float*)d_in[18];
  const float* wg     = (const float*)d_in[19];
  const float* bg     = (const float*)d_in[20];
  const float* un_w   = (const float*)d_in[21];
  const float* un_b   = (const float*)d_in[22];
  const float* cn_w   = (const float*)d_in[23];
  const float* cn_b   = (const float*)d_in[24];
  const float* cl_w   = (const float*)d_in[25];
  const float* cl_b   = (const float*)d_in[26];
  const float* f1w    = (const float*)d_in[27];
  const float* f1b    = (const float*)d_in[28];
  const float* f2w    = (const float*)d_in[29];
  const float* f2b    = (const float*)d_in[30];
  float* out = (float*)d_out;

  float* ws = (float*)d_ws;
  size_t o = 0;
  float* chunk_h   = ws + o; o += (size_t)NB * NCH * DMD;
  float* chunk_max = ws + o; o += (size_t)NB * NCH * DMD;
  float* chunk_cnt = ws + o; o += (size_t)NB * NCH;
  float* chunk_msk = ws + o; o += (size_t)NB * NCH;
  float* smean     = ws + o; o += (size_t)NB * DMD;
  float* smaxb     = ws + o; o += (size_t)NB * DMD;
  float* Dg        = ws + o; o += (size_t)NB * DMD * CDD;
  float* cg        = ws + o; o += (size_t)NB * CDD;
  float* cand_ws   = ws + o; o += (size_t)4 * 16 * 4096;
  float* z_out     = ws + o; o += (size_t)NB * DMD;
  float* err_out   = ws + o; o += (size_t)NB;
  float* gate_out  = ws + o; o += (size_t)NB;
  o = (o + 7) & ~(size_t)7;                 // 32B-align the f16 fragment regions
  _Float16* c1F   = (_Float16*)(ws + o); o += (size_t)16  * 12 * 32 * 16 / 2;
  _Float16* c2F   = (_Float16*)(ws + o); o += (size_t)16  * 20 * 32 * 16 / 2;
  _Float16* opF   = (_Float16*)(ws + o); o += (size_t)8   * 8  * 32 * 16 / 2;
  _Float16* wcduF = (_Float16*)(ws + o); o += (size_t)256 * 4  * 32 * 16 / 2;
  _Float16* wcdcF = (_Float16*)(ws + o); o += (size_t)256 * 1  * 32 * 16 / 2;

  // one-time weight swizzles into WMMA B-fragment layout (f16)
  k_prep<<<(16 * 12 * 32 + 255) / 256, 256, 0, stream>>>(c1w,  c1F,  384, 256);
  k_prep<<<(16 * 20 * 32 + 255) / 256, 256, 0, stream>>>(c2w,  c2F,  640, 256);
  k_prep<<<(8  * 8  * 32 + 255) / 256, 256, 0, stream>>>(op_w, opF,  256, 128);
  k_prep<<<(256 * 4 * 32 + 255) / 256, 256, 0, stream>>>(wcdu, wcduF, 128, 4096);
  k_prep<<<(256 * 1 * 32 + 255) / 256, 256, 0, stream>>>(wcdc, wcdcF, 32, 4096);

  k_init<<<NB, 256, 0, stream>>>(base_D, Dg, cg);
  k_encode<<<NB * (TLEN / 16), 256, 0, stream>>>(tokens, emb, in_w, in_b, c1F, c1b, c2F, c2b,
                                                 opF, op_b, on_w, on_b,
                                                 chunk_h, chunk_cnt, chunk_msk, chunk_max);
  k_surface<<<(NB * DMD + 255) / 256, 256, 0, stream>>>(chunk_h, chunk_cnt, chunk_max, smean, smaxb);
  k_scan<<<4, 256, 0, stream>>>(chunk_h, chunk_msk, wcu, wcc, wcduF, bcdu, wcdcF, bcdc, wg, bg,
                                un_w, un_b, cn_w, cn_b, Dg, cg, cand_ws, z_out, err_out, gate_out);
  k_head<<<NB, 256, 0, stream>>>(z_out, cg, smean, smaxb, err_out, gate_out,
                                 cl_w, cl_b, f1w, f1b, f2w, f2b, out);
}
